// LocalChunkScorer_82583631168048
// MI455X (gfx1250) — matrix operand order repaired and checked by reading, not compiled, not run on hardware
//
#include <hip/hip_runtime.h>
#include <hip/hip_bf16.h>
#include <math.h>

// ---------------- problem constants ----------------
#define BB   8
#define QQ   32
#define HH   256
#define NHH  8
#define JJ   64
#define LCC  128
#define KCTX 8192
#define NLL  2
#define LL   (QQ + LCC)          // 160
#define NN   (BB * JJ)           // 512
#define RR   ((size_t)NN * LL)   // 81920 rows
#define DH   (HH / NHH)          // 32
#define FEAT (3 * HH + 1)        // 769

typedef __attribute__((ext_vector_type(16))) _Float16 v16h;
typedef __attribute__((ext_vector_type(8)))  _Float16 v8h;   // 16 B
typedef __attribute__((ext_vector_type(8)))  float    v8f;
typedef __attribute__((ext_vector_type(4)))  float    v4f;   // 16 B

// ---------------- WMMA helpers (CDNA5 16x16x32 f16 -> f32) ----------------
__device__ __forceinline__ v8f wmma_f16(v16h a, v16h b, v8f c) {
  // (neg_a, A, neg_b, B, c_mod, C, reuse_a, reuse_b)
  return __builtin_amdgcn_wmma_f32_16x16x32_f16(false, a, false, b, (short)0, c,
                                                false, false);
}

__device__ __forceinline__ v8f zero8() {
  v8f z = {0.f, 0.f, 0.f, 0.f, 0.f, 0.f, 0.f, 0.f};
  return z;
}

// A fragment (16x32, M x K) from f16 buffer, row-major, leading dim ld (halfs).
// Layout: lanes 0-15 hold M=lane, K=0..15; lanes 16-31 same M, K=16..31.
// Requires 16-byte alignment of (p + m*ld + k0) runs (all call sites satisfy).
__device__ __forceinline__ v16h frag_A_f16(const _Float16* p, int ld, int m0, int k0) {
  int lane = threadIdx.x & 31;
  const _Float16* row = p + (size_t)(m0 + (lane & 15)) * ld + k0 + ((lane >> 4) << 4);
  v8h lo = *(const v8h*)row;
  v8h hi = *(const v8h*)(row + 8);
  v16h a;
#pragma unroll
  for (int i = 0; i < 8; ++i) {
    a[i] = lo[i];
    a[i + 8] = hi[i];
  }
  return a;
}

// Same but converting from fp32 source on the fly (two 16B + two 16B loads).
__device__ __forceinline__ v16h frag_A_f32(const float* p, int ld, int m0, int k0) {
  int lane = threadIdx.x & 31;
  const float* row = p + (size_t)(m0 + (lane & 15)) * ld + k0 + ((lane >> 4) << 4);
  v4f x0 = *(const v4f*)row;
  v4f x1 = *(const v4f*)(row + 4);
  v4f y0 = *(const v4f*)(row + 8);
  v4f y1 = *(const v4f*)(row + 12);
  v16h a;
#pragma unroll
  for (int i = 0; i < 4; ++i) {
    a[i] = (_Float16)x0[i];
    a[4 + i] = (_Float16)x1[i];
    a[8 + i] = (_Float16)y0[i];
    a[12 + i] = (_Float16)y1[i];
  }
  return a;
}

// B fragment (32x16, K x N) from row-major [K][N] fp32 weights, leading dim ld.
// Layout: lane holds column n = lane&15; K = 16*(lane>>4) + i.
// Scalar strided per lane, but lane-coalesced (16 lanes read one weight row).
__device__ __forceinline__ v16h frag_B_f32(const float* p, int ld, int k0, int n0) {
  int lane = threadIdx.x & 31;
  const float* col = p + (size_t)(k0 + ((lane >> 4) << 4)) * ld + n0 + (lane & 15);
  v16h b;
#pragma unroll
  for (int i = 0; i < 16; ++i) b[i] = (_Float16)col[(size_t)i * ld];
  return b;
}

// B fragment from row-major f16 [K][N].
__device__ __forceinline__ v16h frag_B_f16(const _Float16* p, int ld, int k0, int n0) {
  int lane = threadIdx.x & 31;
  const _Float16* col = p + (size_t)(k0 + ((lane >> 4) << 4)) * ld + n0 + (lane & 15);
  v16h b;
#pragma unroll
  for (int i = 0; i < 16; ++i) b[i] = col[(size_t)i * ld];
  return b;
}

// B = K^T fragment: B[k][n] = Kmat[n0+n][k0+k], Kmat row-major [rows][ld] f16.
// Each lane reads 32 contiguous bytes of one K row -> two 16B loads.
__device__ __forceinline__ v16h frag_Bt_f16(const _Float16* p, int ld, int k0, int n0) {
  int lane = threadIdx.x & 31;
  const _Float16* row = p + (size_t)(n0 + (lane & 15)) * ld + k0 + ((lane >> 4) << 4);
  v8h lo = *(const v8h*)row;
  v8h hi = *(const v8h*)(row + 8);
  v16h b;
#pragma unroll
  for (int i = 0; i < 8; ++i) {
    b[i] = lo[i];
    b[i + 8] = hi[i];
  }
  return b;
}

// ---------------- wave32 reductions ----------------
__device__ __forceinline__ float wave_sum(float v) {
#pragma unroll
  for (int o = 16; o > 0; o >>= 1) v += __shfl_xor(v, o, 32);
  return v;
}
__device__ __forceinline__ float wave_max(float v) {
#pragma unroll
  for (int o = 16; o > 0; o >>= 1) v = fmaxf(v, __shfl_xor(v, o, 32));
  return v;
}

// ---------------- kernel 0: gather question + chunks into seq (fp32) ----------------
__global__ void gather_kernel(const float* __restrict__ qemb,
                              const float* __restrict__ cemb,
                              const int* __restrict__ starts,
                              float* __restrict__ seq) {
  // vectorized: 4 floats per thread-iter
  size_t total = (size_t)NN * LL * (HH / 4);
  for (size_t idx = (size_t)blockIdx.x * blockDim.x + threadIdx.x; idx < total;
       idx += (size_t)gridDim.x * blockDim.x) {
    int c4 = (int)(idx % (HH / 4));
    size_t t = idx / (HH / 4);
    int r = (int)(t % LL);
    int nidx = (int)(t / LL);
    int b = nidx / JJ, j = nidx % JJ;
    v4f v;
    if (r < QQ)
      v = *(const v4f*)(qemb + ((size_t)b * QQ + r) * HH + c4 * 4);
    else
      v = *(const v4f*)(cemb +
                        ((size_t)b * KCTX + starts[b * JJ + j] + (r - QQ)) * HH +
                        c4 * 4);
    *(v4f*)(seq + t * HH + c4 * 4) = v;
  }
}

// ---------------- kernel A: fused QKV projection GEMM ----------------
// seq fp32 [RR][256] x (Wq|Wk|Wv fp32 [256][256]) -> Q,K,V f16 [RR][256]
// block = 32-row tile (2 x 16-row A tiles share every B fragment);
// 8 waves x 6 output col-tiles each; 96 WMMA per wave.
__global__ void __launch_bounds__(256)
qkv_kernel(const float* __restrict__ seq,
           const float* __restrict__ Wq, const float* __restrict__ bq,
           const float* __restrict__ Wk, const float* __restrict__ bk,
           const float* __restrict__ Wv, const float* __restrict__ bv,
           _Float16* __restrict__ Qo, _Float16* __restrict__ Ko,
           _Float16* __restrict__ Vo) {
  int wave = threadIdx.x >> 5, lane = threadIdx.x & 31;
  size_t row0 = (size_t)blockIdx.x * 32;
  const float* A0 = seq + row0 * HH;
  const float* A1 = seq + (row0 + 16) * HH;

  v8f acc[12];
#pragma unroll
  for (int t = 0; t < 12; ++t) acc[t] = zero8();

#pragma unroll
  for (int ks = 0; ks < 8; ++ks) {
    v16h a0 = frag_A_f32(A0, HH, 0, ks * 32);
    v16h a1 = frag_A_f32(A1, HH, 0, ks * 32);
    if (ks < 7) {  // prefetch next k-slice of the activation rows (global_prefetch_b8)
      __builtin_prefetch(A0 + (size_t)(lane & 15) * HH + (ks + 1) * 32, 0, 0);
      __builtin_prefetch(A1 + (size_t)(lane & 15) * HH + (ks + 1) * 32, 0, 0);
    }
#pragma unroll
    for (int t = 0; t < 6; ++t) {
      const float* W = (t < 2) ? Wq : (t < 4) ? Wk : Wv;
      int n0 = (wave + (t & 1) * 8) * 16;
      v16h b = frag_B_f32(W, HH, ks * 32, n0);
      acc[t * 2 + 0] = wmma_f16(a0, b, acc[t * 2 + 0]);
      acc[t * 2 + 1] = wmma_f16(a1, b, acc[t * 2 + 1]);
    }
  }
#pragma unroll
  for (int t = 0; t < 6; ++t) {
    const float* bias = (t < 2) ? bq : (t < 4) ? bk : bv;
    _Float16* dst = (t < 2) ? Qo : (t < 4) ? Ko : Vo;
    int n = (wave + (t & 1) * 8) * 16 + (lane & 15);
    float bb = bias[n];
#pragma unroll
    for (int hhalf = 0; hhalf < 2; ++hhalf) {
#pragma unroll
      for (int r = 0; r < 8; ++r) {
        int m = hhalf * 16 + r + ((lane >> 4) << 3);
        dst[(row0 + m) * HH + n] = (_Float16)(acc[t * 2 + hhalf][r] + bb);
      }
    }
  }
}

// ---------------- kernel B: per-(sequence, head) attention ----------------
// dyn LDS: sQ/sK/sV f16 [160][32] (30720 B) + sS fp32 [160][160] (102400 B)
__global__ void __launch_bounds__(256)
attn_kernel(const _Float16* __restrict__ Qg, const _Float16* __restrict__ Kg,
            const _Float16* __restrict__ Vg, _Float16* __restrict__ Og,
            float* __restrict__ attn_part, int is_last) {
  extern __shared__ char smem[];
  _Float16* sQ = (_Float16*)smem;
  _Float16* sK = sQ + LL * DH;
  _Float16* sV = sK + LL * DH;
  float* sS = (float*)(smem + (size_t)3 * LL * DH * sizeof(_Float16));
  __shared__ float sRed[8];

  int n = blockIdx.x, h = blockIdx.y;
  int tid = threadIdx.x, wave = tid >> 5, lane = tid & 31;
  size_t base = ((size_t)n * LL) * HH + h * DH;

  // vectorized fill: 16B chunks (4 v8h chunks per 32-half row)
  for (int i = tid; i < (LL * DH) / 8; i += 256) {
    int r = i >> 2, c = (i & 3) << 3;
    size_t g = base + (size_t)r * HH + c;
    *(v8h*)(sQ + r * DH + c) = *(const v8h*)(Qg + g);
    *(v8h*)(sK + r * DH + c) = *(const v8h*)(Kg + g);
    *(v8h*)(sV + r * DH + c) = *(const v8h*)(Vg + g);
  }
  __syncthreads();

  // scores S = Q K^T / sqrt(dh): 10x10 tiles, K=32 -> 1 WMMA per tile
  const float scale = 0.17677669529663687f;  // 1/sqrt(32)
  for (int tt = wave; tt < 100; tt += 8) {
    int qt = tt / 10, kt = tt % 10;
    v16h a = frag_A_f16(sQ, DH, qt * 16, 0);
    v16h b = frag_Bt_f16(sK, DH, 0, kt * 16);
    v8f c = wmma_f16(a, b, zero8());
    int nn = kt * 16 + (lane & 15);
#pragma unroll
    for (int r = 0; r < 8; ++r) {
      int m = qt * 16 + r + ((lane >> 4) << 3);
      sS[m * LL + nn] = c[r] * scale;
    }
  }
  __syncthreads();

  // wave-per-row softmax; write probs as f16 in-place (row-aliased, stride 320 halfs)
  float wacc = 0.f;
  for (int row = wave; row < LL; row += 8) {
    float* Sr = sS + row * LL;
    float v[5];
    float mx = -3.4e38f;
#pragma unroll
    for (int i = 0; i < 5; ++i) {
      v[i] = Sr[lane + 32 * i];
      mx = fmaxf(mx, v[i]);
    }
    mx = wave_max(mx);
    float sum = 0.f;
#pragma unroll
    for (int i = 0; i < 5; ++i) {
      v[i] = expf(v[i] - mx);
      sum += v[i];
    }
    sum = wave_sum(sum);
    float inv = 1.f / sum;
    _Float16* Pr = (_Float16*)Sr;  // first 320 B of this row's fp32 storage
    float pmax = 0.f;
#pragma unroll
    for (int i = 0; i < 5; ++i) {
      float p = v[i] * inv;
      Pr[lane + 32 * i] = (_Float16)p;
      if (lane + 32 * i >= QQ) pmax = fmaxf(pmax, p);  // q->context cols only
    }
    if (is_last && row < QQ) {
      pmax = wave_max(pmax);
      if (lane == 0) wacc += pmax;
    }
  }
  if (is_last && lane == 0) sRed[wave] = wacc;
  __syncthreads();
  if (is_last && tid == 0) {
    float s = 0.f;
#pragma unroll
    for (int i = 0; i < 8; ++i) s += sRed[i];
    attn_part[n * NHH + h] = s;  // sum over 32 query rows of row-max
  }

  // O = P @ V : 10x2 tiles, K loop over 160 in steps of 32
  const _Float16* P = (const _Float16*)sS;  // leading dim 2*LL = 320 halfs
  for (int tt = wave; tt < 20; tt += 8) {
    int qt = tt >> 1, ct = tt & 1;
    v8f c = zero8();
#pragma unroll
    for (int ks = 0; ks < 5; ++ks) {
      v16h a = frag_A_f16(P, 2 * LL, qt * 16, ks * 32);
      v16h b = frag_B_f16(sV, DH, ks * 32, ct * 16);
      c = wmma_f16(a, b, c);
    }
    int nn = h * DH + ct * 16 + (lane & 15);
#pragma unroll
    for (int r = 0; r < 8; ++r) {
      int m = qt * 16 + r + ((lane >> 4) << 3);
      Og[((size_t)n * LL + m) * HH + nn] = (_Float16)c[r];
    }
  }
}

// ---------------- kernel C: output projection + residual + LayerNorm ----------------
__global__ void __launch_bounds__(256)
proj_ln_kernel(const _Float16* __restrict__ O, const float* __restrict__ Wo,
               const float* __restrict__ bo, const float* __restrict__ g,
               const float* __restrict__ bta, float* __restrict__ seq) {
  extern __shared__ char smem[];
  float* sAcc = (float*)smem;  // [16][256]
  int wave = threadIdx.x >> 5, lane = threadIdx.x & 31;
  size_t row0 = (size_t)blockIdx.x * 16;
  const _Float16* A = O + row0 * HH;

  // hoist all 8 A fragments once; reuse across this wave's 2 column tiles
  v16h aF[8];
#pragma unroll
  for (int ks = 0; ks < 8; ++ks) aF[ks] = frag_A_f16(A, HH, 0, ks * 32);

  for (int t = wave; t < 16; t += 8) {
    v8f acc = zero8();
#pragma unroll
    for (int ks = 0; ks < 8; ++ks) {
      v16h b = frag_B_f32(Wo, HH, ks * 32, t * 16);
      acc = wmma_f16(aF[ks], b, acc);
    }
    int nn = t * 16 + (lane & 15);
    float bb = bo[nn];
#pragma unroll
    for (int r = 0; r < 8; ++r) {
      int m = r + ((lane >> 4) << 3);
      sAcc[m * HH + nn] = acc[r] + bb + seq[(row0 + m) * HH + nn];
    }
  }
  __syncthreads();

  for (int row = wave; row < 16; row += 8) {
    const float* x = sAcc + row * HH;
    float vals[8];
    float s = 0.f;
#pragma unroll
    for (int i = 0; i < 8; ++i) {
      vals[i] = x[lane + 32 * i];
      s += vals[i];
    }
    float mu = wave_sum(s) * (1.f / 256.f);
    float vs = 0.f;
#pragma unroll
    for (int i = 0; i < 8; ++i) {
      float d = vals[i] - mu;
      vs += d * d;
    }
    float rs = rsqrtf(wave_sum(vs) * (1.f / 256.f) + 1e-5f);
#pragma unroll
    for (int i = 0; i < 8; ++i) {
      int c = lane + 32 * i;
      seq[(row0 + row) * HH + c] = (vals[i] - mu) * rs * g[c] + bta[c];
    }
  }
}

// ---------------- kernel D: pooling + feature + MLP score ----------------
__global__ void __launch_bounds__(256)
score_kernel(const float* __restrict__ seq, const float* __restrict__ attn_part,
             const float* __restrict__ W1, const float* __restrict__ b1,
             const float* __restrict__ W2, const float* __restrict__ b2,
             float* __restrict__ out) {
  __shared__ float feat[FEAT];
  __shared__ float wsum[8];
  int n = blockIdx.x, c = threadIdx.x;
  const float* S = seq + (size_t)n * LL * HH;

  float q = 0.f, cr = 0.f;
  for (int r = 0; r < QQ; ++r) q += S[r * HH + c];
  for (int r = QQ; r < LL; ++r) cr += S[r * HH + c];
  q *= (1.f / QQ);
  cr *= (1.f / LCC);
  feat[c] = q;
  feat[HH + c] = cr;
  feat[2 * HH + c] = q * cr;
  if (c == 0) {
    float s = 0.f;
    for (int h = 0; h < NHH; ++h) s += attn_part[n * NHH + h];
    feat[3 * HH] = s * (1.f / (QQ * NHH));  // mean over heads & query rows
  }
  __syncthreads();

  float a = b1[c];
  for (int f = 0; f < FEAT; ++f) a += feat[f] * W1[f * HH + c];
  a = 0.5f * a * (1.f + erff(a * 0.70710678118654752f));  // exact GELU
  float part = wave_sum(a * W2[c]);
  if ((threadIdx.x & 31) == 0) wsum[threadIdx.x >> 5] = part;
  __syncthreads();
  if (threadIdx.x == 0) {
    float s = b2[0];
#pragma unroll
    for (int i = 0; i < 8; ++i) s += wsum[i];
    out[n] = s;  // n = b*J + j, matches reshape(B, J)
  }
}

// ---------------- host launcher ----------------
extern "C" void kernel_launch(void* const* d_in, const int* in_sizes, int n_in,
                              void* d_out, int out_size, void* d_ws, size_t ws_size,
                              hipStream_t stream) {
  const float* qemb = (const float*)d_in[0];
  const float* cemb = (const float*)d_in[1];
  const int* starts = (const int*)d_in[2];
  // d_in[3] = chunk_len scalar (always 128 per spec)
  const float* Wq = (const float*)d_in[4];
  const float* bq = (const float*)d_in[5];
  const float* Wk = (const float*)d_in[6];
  const float* bk = (const float*)d_in[7];
  const float* Wv = (const float*)d_in[8];
  const float* bv = (const float*)d_in[9];
  const float* Wo = (const float*)d_in[10];
  const float* bo = (const float*)d_in[11];
  const float* lng = (const float*)d_in[12];
  const float* lnb = (const float*)d_in[13];
  const float* W1 = (const float*)d_in[14];
  const float* b1 = (const float*)d_in[15];
  const float* W2 = (const float*)d_in[16];
  const float* b2 = (const float*)d_in[17];
  float* out = (float*)d_out;

  // workspace layout
  char* ws = (char*)d_ws;
  const size_t elems = (size_t)NN * LL * HH;      // 20,971,520
  const size_t seqBytes = elems * sizeof(float);  // 80 MiB
  const size_t hBytes = elems * 2;                // 40 MiB each
  float* seq = (float*)ws;
  _Float16* Qb = (_Float16*)(ws + seqBytes);
  _Float16* Kb = (_Float16*)(ws + seqBytes + hBytes);
  _Float16* Vb = (_Float16*)(ws + seqBytes + 2 * hBytes);
  _Float16* Ob = (_Float16*)(ws + seqBytes + 3 * hBytes);
  float* attn_part = (float*)(ws + seqBytes + 4 * hBytes);  // [512][8]

  gather_kernel<<<2048, 256, 0, stream>>>(qemb, cemb, starts, seq);

  const int qkvBlocks = (int)(RR / 32);   // 2560
  const int lnBlocks = (int)(RR / 16);    // 5120
  const size_t attnLds = (size_t)3 * LL * DH * 2 + (size_t)LL * LL * 4;  // 133,120 B
  for (int l = 0; l < NLL; ++l) {
    const size_t wOff = (size_t)l * HH * HH;
    const size_t bOff = (size_t)l * HH;
    qkv_kernel<<<qkvBlocks, 256, 0, stream>>>(seq, Wq + wOff, bq + bOff,
                                              Wk + wOff, bk + bOff,
                                              Wv + wOff, bv + bOff, Qb, Kb, Vb);
    attn_kernel<<<dim3(NN, NHH), 256, attnLds, stream>>>(Qb, Kb, Vb, Ob, attn_part,
                                                         (l == NLL - 1) ? 1 : 0);
    proj_ln_kernel<<<lnBlocks, 256, 16 * HH * sizeof(float), stream>>>(
        Ob, Wo + wOff, bo + bOff, lng + bOff, lnb + bOff, seq);
  }
  score_kernel<<<NN, 256, 0, stream>>>(seq, attn_part, W1, b1, W2, b2, out);
}
